// GAT_48112223650338
// MI455X (gfx1250) — matrix operand order repaired and checked by reading, not compiled, not run on hardware
//
#include <hip/hip_runtime.h>

// ---------------------------------------------------------------------------
// GAT forward for MI455X (gfx1250): wave32 + v_wmma_f32_16x16x32_f16
// All WMMA operands are contiguous vector loads (B matrices pre-transposed).
// ---------------------------------------------------------------------------

#define DEV __device__ __forceinline__

typedef __attribute__((ext_vector_type(16))) _Float16 v16h;
typedef __attribute__((ext_vector_type(8)))  _Float16 v8h;
typedef __attribute__((ext_vector_type(8)))  float    v8f;

constexpr int BB = 8, NN = 1024, FF = 256, OO = 128, HH = 8, HO = HH * OO;
constexpr float LRELU_A = 0.2f;
constexpr float NEGV    = -9.0e15f;

// A (16x32 f16): lane m = L%16, half = L/16; element e -> K = half*16 + e
// => two contiguous 16B loads per lane per K-chunk.
DEV v16h load_afrag(const _Float16* row, int kbase, int hf) {
  union { v16h v; struct { v8h lo, hi; } p; } u;
  u.p.lo = *(const v8h*)(row + kbase + hf * 16);
  u.p.hi = *(const v8h*)(row + kbase + hf * 16 + 8);
  return u.v;
}

// B (32x16 f16) from a COLUMN-MAJOR (transposed) source: lane holds column col,
// elements are 16 consecutive K values -> one aligned 32B load.
DEV v16h load_bfrag_t(const _Float16* base, size_t col, int ld, int kbase) {
  return *(const v16h*)(base + col * (size_t)ld + kbase);
}

DEV v8f wmma_f16(v16h a, v16h b, v8f c) {
  return __builtin_amdgcn_wmma_f32_16x16x32_f16(false, a, false, b, (short)0, c,
                                                false, false);
}

// ---------------------------- conversion kernels ---------------------------
__global__ void k_f32_to_f16(const float* __restrict__ in, _Float16* __restrict__ out, int n) {
  int i = blockIdx.x * blockDim.x + threadIdx.x;
  if (i < n) out[i] = (_Float16)in[i];
}

// Wt[h][o][k] = W_att[h][k][o]
__global__ void k_make_wt(const float* __restrict__ W, _Float16* __restrict__ wt) {
  int i = blockIdx.x * blockDim.x + threadIdx.x;
  if (i >= HH * OO * FF) return;
  int k = i & (FF - 1), o = (i >> 8) & (OO - 1), hd = i >> 15;
  wt[i] = (_Float16)W[((size_t)hd * FF + k) * OO + o];
}

// wcatT[col][k] : col<128 -> W_out[k][col] ; col>=128 -> lin_w[col-128][k]
__global__ void k_make_wcat(const float* __restrict__ Wout, const float* __restrict__ linw,
                            _Float16* __restrict__ wcat) {
  int i = blockIdx.x * blockDim.x + threadIdx.x;
  if (i >= 256 * HO) return;
  int k = i & (HO - 1), col = i >> 10;
  float v = (col < OO) ? Wout[(size_t)k * OO + col] : linw[(size_t)(col - OO) * HO + k];
  wcat[i] = (_Float16)v;
}

// ------------------- GEMM1: h[b,h] = x[b] @ W[h] (+ fused s1/s2) -----------
// writes hT [b][h][o][n] so attention B-fragments are contiguous in j.
__global__ void __launch_bounds__(32)
k_gemm1(const _Float16* __restrict__ x16, const _Float16* __restrict__ wt,
        const float* __restrict__ asrc, const float* __restrict__ adst,
        _Float16* __restrict__ ht, float* __restrict__ s1, float* __restrict__ s2) {
  __shared__ __align__(32) _Float16 tlo[OO * 16];  // [o][n] tile
  int t = blockIdx.x;
  int itile = t & 63, hd = (t >> 6) & 7, b = t >> 9;
  int lane = threadIdx.x, m = lane & 15, hf = lane >> 4;
  const _Float16* arow = x16 + ((size_t)b * NN + itile * 16 + m) * FF;
  const _Float16* wb   = wt + (size_t)hd * OO * FF;
  v8f acc[8] = {};
  for (int kc = 0; kc < FF / 32; ++kc) {
    v16h a = load_afrag(arow, kc * 32, hf);
#pragma unroll
    for (int ot = 0; ot < 8; ++ot)
      acc[ot] = wmma_f16(a, load_bfrag_t(wb, ot * 16 + m, FF, kc * 32 + hf * 16), acc[ot]);
  }
  // fused score projections: partial over this lane's 8 columns, reduce over m
  float p1[8], p2[8];
#pragma unroll
  for (int r = 0; r < 8; ++r) { p1[r] = 0.f; p2[r] = 0.f; }
#pragma unroll
  for (int ot = 0; ot < 8; ++ot) {
    float va = asrc[(size_t)hd * OO + ot * 16 + m];
    float vd = adst[(size_t)hd * OO + ot * 16 + m];
#pragma unroll
    for (int r = 0; r < 8; ++r) { p1[r] += acc[ot][r] * va; p2[r] += acc[ot][r] * vd; }
  }
#pragma unroll
  for (int d = 1; d < 16; d <<= 1)
#pragma unroll
    for (int r = 0; r < 8; ++r) {
      p1[r] += __shfl_xor(p1[r], d);
      p2[r] += __shfl_xor(p2[r], d);
    }
  if (m == 0) {
    size_t sb = ((size_t)b * HH + hd) * NN + itile * 16 + hf * 8;
#pragma unroll
    for (int r = 0; r < 8; ++r) { s1[sb + r] = p1[r]; s2[sb + r] = p2[r]; }
  }
  // transposed store through LDS -> 32B vector stores of hT
#pragma unroll
  for (int ot = 0; ot < 8; ++ot)
#pragma unroll
    for (int r = 0; r < 8; ++r)
      tlo[(ot * 16 + m) * 16 + r + hf * 8] = (_Float16)acc[ot][r];
  __syncthreads();
  _Float16* hb = ht + ((size_t)b * HH + hd) * OO * NN + itile * 16;
#pragma unroll
  for (int q = 0; q < 4; ++q) {
    int o = q * 32 + lane;
    *(v16h*)(hb + (size_t)o * NN) = *(const v16h*)(tlo + o * 16);
  }
}

// ------- attention layer 0: softmax(mask(lrelu(s1+s2))) @ h, fused ELU -----
__global__ void __launch_bounds__(32)
k_attn1(const _Float16* __restrict__ ht, const float* __restrict__ s1,
        const float* __restrict__ s2, const int* __restrict__ adj,
        _Float16* __restrict__ xcat) {
  __shared__ float s2l[NN];
  int t = blockIdx.x;
  int itile = t & 63, hd = (t >> 6) & 7, b = t >> 9;
  int lane = threadIdx.x, m = lane & 15, hf = lane >> 4;
  const float* s2g = s2 + ((size_t)b * HH + hd) * NN;
  for (int j = lane; j < NN; j += 32) s2l[j] = s2g[j];
  __syncthreads();

  int row = itile * 16 + m;
  float s1m = s1[((size_t)b * HH + hd) * NN + row];
  const int* adjrow = adj + ((size_t)b * NN + row) * NN;

  float mx = -3.4e38f;
  for (int j = hf; j < NN; j += 2) {
    __builtin_prefetch(adjrow + j + 128, 0, 1);
    float e = s1m + s2l[j];
    e = e > 0.f ? e : LRELU_A * e;
    e = adjrow[j] > 0 ? e : NEGV;
    mx = fmaxf(mx, e);
  }
  mx = fmaxf(mx, __shfl_xor(mx, 16));
  float sm = 0.f;
  for (int j = hf; j < NN; j += 2) {
    float e = s1m + s2l[j];
    e = e > 0.f ? e : LRELU_A * e;
    e = adjrow[j] > 0 ? e : NEGV;
    sm += __expf(e - mx);
  }
  sm += __shfl_xor(sm, 16);
  float inv = 1.0f / sm;

  const _Float16* vbase = ht + ((size_t)b * HH + hd) * OO * NN;
  v8f acc[8] = {};
  for (int jc = 0; jc < NN / 32; ++jc) {
    v16h a;
#pragma unroll
    for (int e = 0; e < 16; ++e) {
      int kl = hf * 16 + e;
      int j = jc * 32 + kl;
      float ev = s1m + s2l[j];
      ev = ev > 0.f ? ev : LRELU_A * ev;
      ev = adjrow[j] > 0 ? ev : NEGV;
      a[e] = (_Float16)(__expf(ev - mx) * inv);
    }
#pragma unroll
    for (int ot = 0; ot < 8; ++ot)
      acc[ot] = wmma_f16(a, load_bfrag_t(vbase, ot * 16 + m, NN, jc * 32 + hf * 16), acc[ot]);
  }
#pragma unroll
  for (int ot = 0; ot < 8; ++ot)
#pragma unroll
    for (int r = 0; r < 8; ++r) {
      int mm = r + hf * 8, oo = ot * 16 + m;
      float v = acc[ot][r];
      v = v > 0.f ? v : (__expf(v) - 1.0f);
      xcat[((size_t)b * NN + itile * 16 + mm) * HO + hd * OO + oo] = (_Float16)v;
    }
}

// ---- GEMM2: x@[W_out | lin_w^T]; hsel=0 -> h2T + fused s1b/s2b, 1 -> lin ----
__global__ void __launch_bounds__(32)
k_gemm2(const _Float16* __restrict__ xcat, const _Float16* __restrict__ wcat,
        const float* __restrict__ linb, const float* __restrict__ aosrc,
        const float* __restrict__ aodst, _Float16* __restrict__ h2t,
        float* __restrict__ lin, float* __restrict__ s1b, float* __restrict__ s2b) {
  __shared__ __align__(32) _Float16 tlo[OO * 16];
  int t = blockIdx.x;
  int hsel = t & 1, itile = (t >> 1) & 63, b = t >> 7;
  int lane = threadIdx.x, m = lane & 15, hf = lane >> 4;
  const _Float16* arow = xcat + ((size_t)b * NN + itile * 16 + m) * HO;
  v8f acc[8] = {};
  for (int kc = 0; kc < HO / 32; ++kc) {
    v16h a = load_afrag(arow, kc * 32, hf);
#pragma unroll
    for (int ot = 0; ot < 8; ++ot)
      acc[ot] = wmma_f16(a, load_bfrag_t(wcat, hsel * OO + ot * 16 + m, HO,
                                         kc * 32 + hf * 16), acc[ot]);
  }
  if (hsel == 0) {
    // fused s1b/s2b
    float p1[8], p2[8];
#pragma unroll
    for (int r = 0; r < 8; ++r) { p1[r] = 0.f; p2[r] = 0.f; }
#pragma unroll
    for (int ot = 0; ot < 8; ++ot) {
      float va = aosrc[ot * 16 + m];
      float vd = aodst[ot * 16 + m];
#pragma unroll
      for (int r = 0; r < 8; ++r) { p1[r] += acc[ot][r] * va; p2[r] += acc[ot][r] * vd; }
    }
#pragma unroll
    for (int d = 1; d < 16; d <<= 1)
#pragma unroll
      for (int r = 0; r < 8; ++r) {
        p1[r] += __shfl_xor(p1[r], d);
        p2[r] += __shfl_xor(p2[r], d);
      }
    if (m == 0) {
      size_t sb = (size_t)b * NN + itile * 16 + hf * 8;
#pragma unroll
      for (int r = 0; r < 8; ++r) { s1b[sb + r] = p1[r]; s2b[sb + r] = p2[r]; }
    }
    // transposed store of h2T
#pragma unroll
    for (int ot = 0; ot < 8; ++ot)
#pragma unroll
      for (int r = 0; r < 8; ++r)
        tlo[(ot * 16 + m) * 16 + r + hf * 8] = (_Float16)acc[ot][r];
    __syncthreads();
    _Float16* hb = h2t + (size_t)b * OO * NN + itile * 16;
#pragma unroll
    for (int q = 0; q < 4; ++q) {
      int o = q * 32 + lane;
      *(v16h*)(hb + (size_t)o * NN) = *(const v16h*)(tlo + o * 16);
    }
  } else {
#pragma unroll
    for (int ot = 0; ot < 8; ++ot)
#pragma unroll
      for (int r = 0; r < 8; ++r) {
        int mm = itile * 16 + r + hf * 8, oo = ot * 16 + m;
        lin[((size_t)b * NN + mm) * OO + oo] = acc[ot][r] + linb[oo];
      }
  }
}

// ---- attention layer 1 + residual + final projection + ReLU, all fused ----
__global__ void __launch_bounds__(32)
k_attn2(const _Float16* __restrict__ h2t, const float* __restrict__ s1,
        const float* __restrict__ s2, const int* __restrict__ adj,
        const float* __restrict__ lin, const _Float16* __restrict__ lnw,
        const float* __restrict__ lnb, float* __restrict__ out) {
  __shared__ float s2l[NN];
  __shared__ __align__(32) _Float16 tl[16 * OO];
  int t = blockIdx.x;
  int itile = t & 63, b = t >> 6;
  int lane = threadIdx.x, m = lane & 15, hf = lane >> 4;
  const float* s2g = s2 + (size_t)b * NN;
  for (int j = lane; j < NN; j += 32) s2l[j] = s2g[j];
  __syncthreads();

  int row = itile * 16 + m;
  float s1m = s1[(size_t)b * NN + row];
  const int* adjrow = adj + ((size_t)b * NN + row) * NN;

  float mx = -3.4e38f;
  for (int j = hf; j < NN; j += 2) {
    __builtin_prefetch(adjrow + j + 128, 0, 1);
    float e = s1m + s2l[j];
    e = e > 0.f ? e : LRELU_A * e;
    e = adjrow[j] > 0 ? e : NEGV;
    mx = fmaxf(mx, e);
  }
  mx = fmaxf(mx, __shfl_xor(mx, 16));
  float sm = 0.f;
  for (int j = hf; j < NN; j += 2) {
    float e = s1m + s2l[j];
    e = e > 0.f ? e : LRELU_A * e;
    e = adjrow[j] > 0 ? e : NEGV;
    sm += __expf(e - mx);
  }
  sm += __shfl_xor(sm, 16);
  float inv = 1.0f / sm;

  const _Float16* vbase = h2t + (size_t)b * OO * NN;
  v8f acc[8] = {};
  for (int jc = 0; jc < NN / 32; ++jc) {
    v16h a;
#pragma unroll
    for (int e = 0; e < 16; ++e) {
      int kl = hf * 16 + e;
      int j = jc * 32 + kl;
      float ev = s1m + s2l[j];
      ev = ev > 0.f ? ev : LRELU_A * ev;
      ev = adjrow[j] > 0 ? ev : NEGV;
      a[e] = (_Float16)(__expf(ev - mx) * inv);
    }
#pragma unroll
    for (int ot = 0; ot < 8; ++ot)
      acc[ot] = wmma_f16(a, load_bfrag_t(vbase, ot * 16 + m, NN, jc * 32 + hf * 16), acc[ot]);
  }
  // residual add, stage t-tile in LDS for the final projection
#pragma unroll
  for (int ot = 0; ot < 8; ++ot)
#pragma unroll
    for (int r = 0; r < 8; ++r) {
      int mm = r + hf * 8, oo = ot * 16 + m;
      float v = acc[ot][r] + lin[((size_t)b * NN + itile * 16 + mm) * OO + oo];
      tl[mm * OO + oo] = (_Float16)v;
    }
  __syncthreads();

  // final: out = relu(t @ ln_w^T + ln_b); ln_w row-major IS the transposed B.
  v8f acc2[8] = {};
  const _Float16* trow = tl + m * OO;
  for (int kc = 0; kc < OO / 32; ++kc) {
    v16h a = load_afrag(trow, kc * 32, hf);
#pragma unroll
    for (int ot = 0; ot < 8; ++ot)
      acc2[ot] = wmma_f16(a, load_bfrag_t(lnw, ot * 16 + m, OO, kc * 32 + hf * 16), acc2[ot]);
  }
#pragma unroll
  for (int ot = 0; ot < 8; ++ot)
#pragma unroll
    for (int r = 0; r < 8; ++r) {
      int mm = itile * 16 + r + hf * 8, oo = ot * 16 + m;
      float v = acc2[ot][r] + lnb[oo];
      out[((size_t)b * NN + mm) * OO + oo] = v > 0.f ? v : 0.f;
    }
}

// ------------------------------- host side ---------------------------------
extern "C" void kernel_launch(void* const* d_in, const int* in_sizes, int n_in,
                              void* d_out, int out_size, void* d_ws, size_t ws_size,
                              hipStream_t stream) {
  const float* inputs = (const float*)d_in[0];
  const int*   adj    = (const int*)d_in[1];
  /* d_in[2] mask: unused by forward */
  const float* W_att  = (const float*)d_in[3];
  const float* a_src  = (const float*)d_in[4];
  const float* a_dst  = (const float*)d_in[5];
  const float* W_out  = (const float*)d_in[6];
  const float* ao_src = (const float*)d_in[7];
  const float* ao_dst = (const float*)d_in[8];
  const float* lin_w  = (const float*)d_in[9];
  const float* lin_b  = (const float*)d_in[10];
  const float* ln_w   = (const float*)d_in[11];
  const float* ln_b   = (const float*)d_in[12];
  float* out = (float*)d_out;

  char* ws = (char*)d_ws;
  size_t off = 0;
  auto carve = [&](size_t bytes) {
    void* p = ws + off;
    off += (bytes + 255) & ~(size_t)255;
    return p;
  };
  _Float16* x16   = (_Float16*)carve((size_t)BB * NN * FF * 2);      // 4 MB
  _Float16* wt16  = (_Float16*)carve((size_t)HH * OO * FF * 2);      // 0.5 MB
  _Float16* ht16  = (_Float16*)carve((size_t)BB * HH * OO * NN * 2); // 16 MB
  float*    s1    = (float*)carve((size_t)BB * HH * NN * 4);
  float*    s2    = (float*)carve((size_t)BB * HH * NN * 4);
  _Float16* xcat  = (_Float16*)carve((size_t)BB * NN * HO * 2);      // 16 MB
  _Float16* wcat  = (_Float16*)carve((size_t)256 * HO * 2);
  _Float16* h2t   = (_Float16*)carve((size_t)BB * OO * NN * 2);
  float*    lin   = (float*)carve((size_t)BB * NN * OO * 4);
  float*    s1b   = (float*)carve((size_t)BB * NN * 4);
  float*    s2b   = (float*)carve((size_t)BB * NN * 4);
  _Float16* lnw16 = (_Float16*)carve((size_t)OO * OO * 2);

  {
    int n = BB * NN * FF;
    k_f32_to_f16<<<(n + 255) / 256, 256, 0, stream>>>(inputs, x16, n);
  }
  {
    int n = HH * OO * FF;
    k_make_wt<<<(n + 255) / 256, 256, 0, stream>>>(W_att, wt16);
  }
  {
    int n = 256 * HO;
    k_make_wcat<<<(n + 255) / 256, 256, 0, stream>>>(W_out, lin_w, wcat);
  }
  {
    int n = OO * OO;
    k_f32_to_f16<<<(n + 255) / 256, 256, 0, stream>>>(ln_w, lnw16, n);
  }

  // layer 0
  k_gemm1<<<BB * HH * (NN / 16), 32, 0, stream>>>(x16, wt16, a_src, a_dst, ht16, s1, s2);
  k_attn1<<<BB * HH * (NN / 16), 32, 0, stream>>>(ht16, s1, s2, adj, xcat);

  // layer 1
  k_gemm2<<<BB * (NN / 16) * 2, 32, 0, stream>>>(xcat, wcat, lin_b, ao_src, ao_dst,
                                                 h2t, lin, s1b, s2b);
  k_attn2<<<BB * (NN / 16), 32, 0, stream>>>(h2t, s1b, s2b, adj, lin, lnw16, ln_b, out);
}
